// Net_29575144800872
// MI455X (gfx1250) — matrix operand order repaired and checked by reading, not compiled, not run on hardware
//
#include <hip/hip_runtime.h>
#include <hip/hip_bf16.h>

// ---- problem constants (from reference) ----
#define BB 512   // batch
#define VV 256   // vocab
#define MM 128   // memory_size
#define HH 512   // hidden
#define TT 16    // timing bits
#define K1 (MM*VV + MM*TT)   // 34816, inner dim of GEMM1

typedef __bf16 bf16;
typedef __attribute__((ext_vector_type(16))) __bf16 v16bf;
typedef __attribute__((ext_vector_type(8)))  __bf16 v8bf;
typedef __attribute__((ext_vector_type(8)))  float  v8f;
typedef __attribute__((ext_vector_type(4)))  unsigned int v4u;
typedef __attribute__((ext_vector_type(8)))  int v8i;
typedef __attribute__((ext_vector_type(4)))  int v4i;

union BF16x16 { v16bf v; v8bf h[2]; };

#if defined(__has_builtin)
#if __has_builtin(__builtin_amdgcn_tensor_load_to_lds)
#define HAVE_TDM 1
#endif
#endif
#ifndef HAVE_TDM
#define HAVE_TDM 0
#endif

#if __has_include(<hip/amd_detail/amd_gfx1250_TDM.h>)
#define TDM_SIX_ARG 1
#else
#define TDM_SIX_ARG 0
#endif

// ---------------------------------------------------------------------------
// Kernel 1: per-batch argmin(memory_surprise) (first-min), timing update,
// stable O(M^2) rank -> order[], sorted_t[].   1 block per batch, 128 threads.
// (surprise itself is dead code: it only feeds msurp, which is never read;
//  argmin is invariant under the *ALPHA scale.)
// ---------------------------------------------------------------------------
__global__ void prep_kernel(const float* __restrict__ msurp,
                            const int*   __restrict__ mtim,
                            int* __restrict__ idx_out,
                            int* __restrict__ order,
                            int* __restrict__ sorted_t) {
    int b = blockIdx.x;
    int m = threadIdx.x;
    __shared__ float s_val[MM];
    __shared__ int   s_idx[MM];
    __shared__ int   s_t[MM];

    s_val[m] = msurp[b * MM + m];
    s_idx[m] = m;
    __syncthreads();
    for (int off = MM / 2; off > 0; off >>= 1) {
        if (m < off) {
            float v2 = s_val[m + off]; int i2 = s_idx[m + off];
            if (v2 < s_val[m] || (v2 == s_val[m] && i2 < s_idx[m])) {
                s_val[m] = v2; s_idx[m] = i2;
            }
        }
        __syncthreads();
    }
    int idx = s_idx[0];
    int t = (m == idx) ? 0 : (mtim[b * MM + m] + 1);
    s_t[m] = t;
    __syncthreads();
    int rank = 0;
    for (int k = 0; k < MM; ++k) {
        int tk = s_t[k];
        rank += (tk < t) || (tk == t && k < m);
    }
    order[b * MM + rank]    = m;
    sorted_t[b * MM + rank] = t;
    if (m == 0) idx_out[b] = idx;
}

// ---------------------------------------------------------------------------
// Kernel 2: materialize pred_in in bf16: [B, K1]
// ---------------------------------------------------------------------------
__global__ void build_predin(const float* __restrict__ x,
                             const float* __restrict__ memory,
                             const int*   __restrict__ idxb,
                             const int*   __restrict__ order,
                             const int*   __restrict__ sorted_t,
                             bf16* __restrict__ predin) {
    int m = blockIdx.x, b = blockIdx.y, v = threadIdx.x;
    int row = order[b * MM + m];
    float val = (row == idxb[b]) ? x[b * VV + v]
                                 : memory[((size_t)b * MM + row) * VV + v];
    predin[(size_t)b * K1 + m * VV + v] = (bf16)val;
    if (v < TT) {
        int bit = (sorted_t[b * MM + m] >> v) & 1;
        predin[(size_t)b * K1 + MM * VV + m * TT + v] = (bf16)(float)bit;
    }
}

// ---------------------------------------------------------------------------
// Kernel 3: LDS-tiled transpose + fp32->bf16: in[R,C] row-major -> out[C,R]
// ---------------------------------------------------------------------------
__global__ void transpose_to_bf16(const float* __restrict__ in,
                                  bf16* __restrict__ out, int R, int C) {
    __shared__ float tile[32][33];
    int c0 = blockIdx.x * 32, r0 = blockIdx.y * 32;
    for (int i = threadIdx.y; i < 32; i += 8) {
        int r = r0 + i, c = c0 + threadIdx.x;
        tile[i][threadIdx.x] = (r < R && c < C) ? in[(size_t)r * C + c] : 0.f;
    }
    __syncthreads();
    for (int i = threadIdx.y; i < 32; i += 8) {
        int c = c0 + i, r = r0 + threadIdx.x;
        if (c < C && r < R) out[(size_t)c * R + r] = (bf16)tile[threadIdx.x][i];
    }
}

// ---------------------------------------------------------------------------
// TDM: stage a 2D tile (rowsTile rows x rowElems bf16) from global into LDS,
// contiguous row-major in LDS. D# built per CDNA5 ISA ch.8.
// ---------------------------------------------------------------------------
__device__ __forceinline__ unsigned lds_off32(const void* p) {
    // generic LDS address: addr[31:0] is the workgroup-relative LDS offset
    return (unsigned)(size_t)p;
}

__device__ __forceinline__ void tdm_load_2d(unsigned ldsAddr, const bf16* gsrc,
                                            unsigned rowElems, unsigned rowsTile,
                                            unsigned tensorDim0, unsigned tensorDim1,
                                            unsigned rowStrideElems) {
#if HAVE_TDM
    unsigned long long ga = (unsigned long long)(size_t)gsrc;
    v4u g0;
    g0[0] = 1u;                                         // count=1, user mode, no gather
    g0[1] = ldsAddr;                                    // lds_addr (bytes)
    g0[2] = (unsigned)(ga & 0xFFFFFFFFu);               // global_addr[31:0]
    g0[3] = (unsigned)((ga >> 32) & 0x01FFFFFFu)        // global_addr[56:32]
          | 0x80000000u;                                // type=2 ("image")
    v8i g1;
    g1[0] = (int)(1u << 16);                            // data_size=1 -> 2 bytes/elem
    g1[1] = (int)((tensorDim0 & 0xFFFFu) << 16);        // tensor_dim0[15:0] @ bits63:48
    g1[2] = (int)((tensorDim0 >> 16) & 0xFFFFu)         // tensor_dim0[31:16]
          | (int)((tensorDim1 & 0xFFFFu) << 16);        // tensor_dim1[15:0]
    g1[3] = (int)((tensorDim1 >> 16) & 0xFFFFu)         // tensor_dim1[31:16]
          | (int)((rowElems & 0xFFFFu) << 16);          // tile_dim0
    g1[4] = (int)(rowsTile & 0xFFFFu);                  // tile_dim1 (tile_dim2=0)
    g1[5] = (int)rowStrideElems;                        // tensor_dim0_stride[31:0]
    g1[6] = 0;                                          // stride[47:32], dim1_stride lo
    g1[7] = 0;
    v4i z4 = {0, 0, 0, 0};
#if TDM_SIX_ARG
    v8i z8 = {0, 0, 0, 0, 0, 0, 0, 0};
    __builtin_amdgcn_tensor_load_to_lds(g0, g1, z4, z4, z8, 0);
#else
    __builtin_amdgcn_tensor_load_to_lds(g0, g1, z4, z4, 0);
#endif
#else
    // cooperative fallback (wave 0, 32 lanes): copy rowsTile rows of rowElems bf16
    bf16* dst = (bf16*)(size_t)ldsAddr;  // unused path on TDM-capable toolchains
    int lane = threadIdx.x & 31;
    int segs = (int)(rowsTile * rowElems) / 8;
    for (int i = lane; i < segs; i += 32) {
        int row = (i * 8) / (int)rowElems;
        int col = (i * 8) % (int)rowElems;
        *(v8bf*)(dst + i * 8) = *(const v8bf*)(gsrc + (size_t)row * rowStrideElems + col);
    }
#endif
}

__device__ __forceinline__ v8f wmma_bf16(const BF16x16& a, const BF16x16& b, v8f c) {
    return __builtin_amdgcn_wmma_f32_16x16x32_bf16(
        /*neg_a=*/false, a.v, /*neg_b=*/false, b.v,
        /*c_mod=*/(short)0, c, /*reuse_a=*/false, /*reuse_b=*/false);
}

// ---------------------------------------------------------------------------
// Kernel 4: WMMA bf16 GEMM, out[Md,Nd] = A[Md,Kd] * Bt[Nd,Kd]^T + bias
//  * block = 8 waves, block tile = 256(M) x 32(N); wave tile = 32x32 (2x2)
//  * B panel (32 rows x 32 K bf16 = 2 KB) staged into LDS by TDM, double-
//    buffered, synchronized with s_wait_tensorcnt + workgroup barrier
//  * A rows are disjoint across waves -> direct global b128 loads
// ---------------------------------------------------------------------------
template <int RELU, int WRITE_F32>
__global__ void gemm_wmma_tdm(const bf16* __restrict__ A,
                              const bf16* __restrict__ Bt,
                              const float* __restrict__ bias,
                              float* __restrict__ outF,
                              bf16*  __restrict__ outB,
                              int Md, int Nd, int Kd) {
    __shared__ __align__(128) bf16 sB[2][32 * 32];   // 2 x 2KB double buffer

    const int wave = threadIdx.x >> 5;
    const int lane = threadIdx.x & 31;
    const int half = lane >> 4;          // ISA A/B half: lanes 0-15 vs 16-31
    const int lr   = lane & 15;

    const int m0 = blockIdx.x * 256 + wave * 32;     // wave's 32 output rows
    const int n0 = blockIdx.y * 32;                  // block's 32 output cols

    const bf16* pA0 = A + (size_t)(m0 + lr) * Kd + half * 8;        // M-tile 0
    const bf16* pA1 = A + (size_t)(m0 + 16 + lr) * Kd + half * 8;   // M-tile 1
    const bf16* panel = Bt + (size_t)n0 * Kd;

    // prologue: stage K-chunk 0
    if (wave == 0)
        tdm_load_2d(lds_off32(&sB[0][0]), panel, 32, 32, Kd, Nd, Kd);
    __builtin_amdgcn_s_wait_tensorcnt((short)0);
    __syncthreads();

    v8f c00 = {}, c01 = {}, c10 = {}, c11 = {};
    const int nch = Kd >> 5;
    for (int kc = 0; kc < nch; ++kc) {
        const int cur = kc & 1;
        if (kc + 1 < nch && wave == 0)   // wave-uniform branch: EXEC stays all-1
            tdm_load_2d(lds_off32(&sB[cur ^ 1][0]), panel + (size_t)(kc + 1) * 32,
                        32, 32, Kd, Nd, Kd);

        const int k0 = kc * 32;
        BF16x16 a0, a1, b0, b1;
        a0.h[0] = *(const v8bf*)(pA0 + k0);
        a0.h[1] = *(const v8bf*)(pA0 + k0 + 16);
        a1.h[0] = *(const v8bf*)(pA1 + k0);
        a1.h[1] = *(const v8bf*)(pA1 + k0 + 16);
        const bf16* bb = &sB[cur][0];
        b0.h[0] = *(const v8bf*)(bb + lr * 32 + half * 8);          // cols n0+0..15
        b0.h[1] = *(const v8bf*)(bb + lr * 32 + half * 8 + 16);
        b1.h[0] = *(const v8bf*)(bb + (16 + lr) * 32 + half * 8);   // cols n0+16..31
        b1.h[1] = *(const v8bf*)(bb + (16 + lr) * 32 + half * 8 + 16);

        c00 = wmma_bf16(a0, b0, c00);
        c01 = wmma_bf16(a0, b1, c01);
        c10 = wmma_bf16(a1, b0, c10);
        c11 = wmma_bf16(a1, b1, c11);

        __builtin_amdgcn_s_wait_tensorcnt((short)0);  // next buffer landed
        __syncthreads();                              // all waves done with cur
    }

    // epilogue: bias + (relu) + store 2x2 tiles
    const float bv0 = bias[n0 + lr];
    const float bv1 = bias[n0 + 16 + lr];
#pragma unroll
    for (int r = 0; r < 8; ++r) {
        int row0 = m0 + r + half * 8;
        int row1 = m0 + 16 + r + half * 8;
        float v00 = c00[r] + bv0, v01 = c01[r] + bv1;
        float v10 = c10[r] + bv0, v11 = c11[r] + bv1;
        if (RELU) {
            v00 = v00 > 0.f ? v00 : 0.f;  v01 = v01 > 0.f ? v01 : 0.f;
            v10 = v10 > 0.f ? v10 : 0.f;  v11 = v11 > 0.f ? v11 : 0.f;
        }
        if (WRITE_F32) {
            outF[(size_t)row0 * Nd + n0 + lr]      = v00;
            outF[(size_t)row0 * Nd + n0 + 16 + lr] = v01;
            outF[(size_t)row1 * Nd + n0 + lr]      = v10;
            outF[(size_t)row1 * Nd + n0 + 16 + lr] = v11;
        } else {
            outB[(size_t)row0 * Nd + n0 + lr]      = (bf16)v00;
            outB[(size_t)row0 * Nd + n0 + 16 + lr] = (bf16)v01;
            outB[(size_t)row1 * Nd + n0 + lr]      = (bf16)v10;
            outB[(size_t)row1 * Nd + n0 + 16 + lr] = (bf16)v11;
        }
    }
}

// ---------------------------------------------------------------------------
extern "C" void kernel_launch(void* const* d_in, const int* in_sizes, int n_in,
                              void* d_out, int out_size, void* d_ws, size_t ws_size,
                              hipStream_t stream) {
    const float* x      = (const float*)d_in[0];
    const float* memory = (const float*)d_in[1];
    const int*   mtim   = (const int*)  d_in[2];
    const float* msurp  = (const float*)d_in[3];
    // d_in[4] last_prediction: dead (surprise only feeds msurp, never read)
    const float* W1     = (const float*)d_in[5];
    const float* b1     = (const float*)d_in[6];
    const float* W2     = (const float*)d_in[7];
    const float* b2     = (const float*)d_in[8];
    float* out = (float*)d_out;

    char* ws = (char*)d_ws;
    const size_t SZ_PREDIN = (size_t)BB * K1 * sizeof(bf16);
    const size_t SZ_W1T    = (size_t)HH * K1 * sizeof(bf16);
    const size_t SZ_HWS    = (size_t)BB * HH * sizeof(bf16);
    const size_t SZ_W2T    = (size_t)VV * HH * sizeof(bf16);
    const size_t SZ_IDX    = (size_t)BB * sizeof(int);
    const size_t SZ_ORD    = (size_t)BB * MM * sizeof(int);

    size_t off = 0;
    bf16* predin  = (bf16*)(ws + off); off += SZ_PREDIN;
    bf16* w1t     = (bf16*)(ws + off); off += SZ_W1T;
    bf16* hws     = (bf16*)(ws + off); off += SZ_HWS;
    bf16* w2t     = (bf16*)(ws + off); off += SZ_W2T;
    int*  idxb    = (int*) (ws + off); off += ((SZ_IDX + 255) & ~255ull);
    int*  order   = (int*) (ws + off); off += SZ_ORD;
    int*  sortedt = (int*) (ws + off); off += SZ_ORD;

    // 1) argmin + timing update + stable rank
    prep_kernel<<<BB, MM, 0, stream>>>(msurp, mtim, idxb, order, sortedt);

    // 2) materialize pred_in (bf16)
    build_predin<<<dim3(MM, BB), VV, 0, stream>>>(x, memory, idxb, order, sortedt, predin);

    // 3) W1 -> W1^T (bf16), W2 -> W2^T (bf16)
    transpose_to_bf16<<<dim3(HH / 32, K1 / 32), dim3(32, 8), 0, stream>>>(W1, w1t, K1, HH);
    transpose_to_bf16<<<dim3(VV / 32, HH / 32), dim3(32, 8), 0, stream>>>(W2, w2t, HH, VV);

    // 4) h = relu(pred_in @ W1 + b1) -> bf16   (blocks: 2 x 16, 8 waves each)
    gemm_wmma_tdm<1, 0><<<dim3(BB / 256, HH / 32), 256, 0, stream>>>(
        predin, w1t, b1, nullptr, hws, BB, HH, K1);

    // 5) out = h @ W2 + b2 -> fp32             (blocks: 2 x 8)
    gemm_wmma_tdm<0, 1><<<dim3(BB / 256, VV / 32), 256, 0, stream>>>(
        hws, w2t, b2, out, nullptr, BB, VV, HH);
}
